// MutiheadRelativeAttention_6356551598440
// MI455X (gfx1250) — matrix operand order repaired and checked by reading, not compiled, not run on hardware
//
#include <hip/hip_runtime.h>

// ---------------------------------------------------------------------------
// MultiheadRelativeAttention on gfx1250 (MI455X), all GEMMs on
// v_wmma_f32_16x16x32_f16 (f16 operands, f32 accumulate).
//
// B=4, T=1024, E=1024, H=16, D=64, P=257 (padded to 288 for WMMA K-tiling).
// Pair index p = b*16+h (64 pairs). Torch-reshape quirk permutations:
//   scores at pair p use qdot from pair m  = (p%16)*4 + p/16
//   w2 of attn pair p lands at output pair p2 = (p%4)*16 + p/4
// ---------------------------------------------------------------------------

typedef __attribute__((ext_vector_type(16))) _Float16 v16h;
typedef __attribute__((ext_vector_type(8)))  float    v8f;

#define WMMA_F16(a, b, c) \
  __builtin_amdgcn_wmma_f32_16x16x32_f16(false, (a), false, (b), (short)0, (c), false, false)

// ---- fragment loaders (ISA 7.12.2 layouts, wave32) -------------------------

// A-matrix 16x32 f16: lane L (grp=L>>4, m=L&15) holds row m.
// elems 0..7  -> K = grp*8 + e          (0..7 / 8..15)
// elems 8..15 -> K = 16 + grp*8 + (e-8) (16..23 / 24..31)
__device__ inline v16h load_A_f16(const _Float16* base, int ldr, int m0, int k0) {
  int lane = threadIdx.x & 31;
  int grp = lane >> 4, m = lane & 15;
  const _Float16* row = base + (size_t)(m0 + m) * ldr + k0;
  v16h a;
#pragma unroll
  for (int e = 0; e < 8; ++e) a[e] = row[grp * 8 + e];
#pragma unroll
  for (int e = 0; e < 8; ++e) a[8 + e] = row[16 + grp * 8 + e];
  return a;
}

// B-matrix 32x16 f16 where source is stored as src[col][contraction]
// (i.e. B^T row-major): lane n (col), elems e -> K = grp*8 + e + 8*(e>>3) (table order).
__device__ inline v16h load_B_f16(const _Float16* base, int ldr, int n0, int k0) {
  int lane = threadIdx.x & 31;
  int grp = lane >> 4, n = lane & 15;
  const _Float16* row = base + (size_t)(n0 + n) * ldr + k0 + grp * 8;
  v16h b;
#pragma unroll
  for (int e = 0; e < 16; ++e) b[e] = row[e + (e >> 3) * 8];
  return b;
}

// B-matrix 32x16 f16 where source is stored as src[contraction][col]
// (row-major B): strided gather.
__device__ inline v16h load_B_f16_rm(const _Float16* base, int ldr, int k0, int n0) {
  int lane = threadIdx.x & 31;
  int grp = lane >> 4, n = lane & 15;
  v16h b;
#pragma unroll
  for (int e = 0; e < 16; ++e) b[e] = base[(size_t)(k0 + grp * 8 + e + (e >> 3) * 8) * ldr + n0 + n];
  return b;
}

// ---- K0: f32 -> f16 conversions -------------------------------------------

__global__ void cvt_f32_f16(const float* __restrict__ src, _Float16* __restrict__ dst, int n) {
  int i = blockIdx.x * blockDim.x + threadIdx.x;
  if (i < n) dst[i] = (_Float16)src[i];
}

__global__ void cvt_pad_table(const float* __restrict__ src, _Float16* __restrict__ dst,
                              int rows_src, int rows_dst, int cols) {
  int i = blockIdx.x * blockDim.x + threadIdx.x;
  if (i < rows_dst * cols) {
    int r = i / cols;
    dst[i] = (r < rows_src) ? (_Float16)src[i] : (_Float16)0.0f;
  }
}

// ---- K1: Q/K/V projection: y = hh @ W^T + b, scattered to (pair,t,d) f16 ---
// grid (128, 16, 3) block 256 (8 waves); wave -> one 16x16 C tile.

__global__ __launch_bounds__(256) void proj_kernel(
    const _Float16* __restrict__ hh,
    const _Float16* __restrict__ wq, const _Float16* __restrict__ wk, const _Float16* __restrict__ wv,
    const float* __restrict__ bq, const float* __restrict__ bk, const float* __restrict__ bv,
    _Float16* __restrict__ qb, _Float16* __restrict__ kb, _Float16* __restrict__ vb) {
  const _Float16* W;
  const float* bias;
  _Float16* dst;
  if (blockIdx.z == 0)      { W = wq; bias = bq; dst = qb; }
  else if (blockIdx.z == 1) { W = wk; bias = bk; dst = kb; }
  else                      { W = wv; bias = bv; dst = vb; }

  int wave = threadIdx.x >> 5;
  int i0 = blockIdx.x * 32 + (wave >> 2) * 16;
  int j0 = blockIdx.y * 64 + (wave & 3) * 16;

  v8f c = {};
  for (int e0 = 0; e0 < 1024; e0 += 32) {
    v16h a = load_A_f16(hh, 1024, i0, e0);
    v16h b = load_B_f16(W, 1024, j0, e0);
    c = WMMA_F16(a, b, c);
  }
  int lane = threadIdx.x & 31, grp = lane >> 4, N = lane & 15;
  int j = j0 + N, h = j >> 6, d = j & 63;
#pragma unroll
  for (int r = 0; r < 8; ++r) {
    int i = i0 + r + 8 * grp;
    int bb = i >> 10, t = i & 1023;
    dst[(((size_t)(bb * 16 + h) * 1024) + t) * 64 + d] = (_Float16)(c[r] + bias[j]);
  }
}

// ---- K2: qdot[m][t][j] = qh_m[t] . rel_k[j]  (f32 out, j padded to 288) ----
// grid (64, 18, 64) block 32.

__global__ __launch_bounds__(32) void qdot_kernel(const _Float16* __restrict__ qb,
                                                  const _Float16* __restrict__ relk_h,
                                                  float* __restrict__ qdot) {
  int m = blockIdx.z, t0 = blockIdx.x * 16, j0 = blockIdx.y * 16;
  const _Float16* qm = qb + (size_t)m * 1024 * 64;
  v8f c = {};
  c = WMMA_F16(load_A_f16(qm, 64, t0, 0),  load_B_f16(relk_h, 64, j0, 0),  c);
  c = WMMA_F16(load_A_f16(qm, 64, t0, 32), load_B_f16(relk_h, 64, j0, 32), c);
  int lane = threadIdx.x & 31, grp = lane >> 4, N = lane & 15;
  float* out = qdot + ((size_t)m * 1024 + t0) * 288 + j0;
#pragma unroll
  for (int r = 0; r < 8; ++r) out[(size_t)(r + 8 * grp) * 288 + N] = c[r];
}

// ---- K3: attention: scores + softmax + w1 = P.V + w2 = pw.rel_v ------------
// One wave per block; block = (q-tile of 16 rows, pair p). grid (64, 64).

__global__ __launch_bounds__(32) void attn_kernel(
    const _Float16* __restrict__ qb, const _Float16* __restrict__ kb,
    const _Float16* __restrict__ vb, const float* __restrict__ qdot,
    const _Float16* __restrict__ relv_h,
    float* __restrict__ w1buf, float* __restrict__ w2buf) {
  __shared__ _Float16 Pb[16][1024];   // score row block, then exp() values
  __shared__ float    qdl[16][288];   // staged qdot rows (permuted pair m)
  __shared__ _Float16 pwb[16][288];   // relative-distance buckets (normalized)
  __shared__ float    rmax[16];
  __shared__ float    rsum[16];

  int q0 = blockIdx.x * 16;
  int p  = blockIdx.y;
  int lane = threadIdx.x & 31, grp = lane >> 4, N = lane & 15;
  int m  = (p & 15) * 4 + (p >> 4);   // qdot source pair
  int p2 = (p & 3) * 16 + (p >> 2);   // w2 destination pair

  // stage qdot rows q0..q0+15 of pair m
  {
    const float* qd = qdot + ((size_t)m * 1024 + q0) * 288;
    float* dst = &qdl[0][0];
    for (int idx = lane; idx < 16 * 288; idx += 32) dst[idx] = qd[idx];
  }

  const _Float16* qp = qb + (size_t)p * 65536;
  const _Float16* kp = kb + (size_t)p * 65536;
  const _Float16* vp = vb + (size_t)p * 65536;

  v16h qa0 = load_A_f16(qp, 64, q0, 0);
  v16h qa1 = load_A_f16(qp, 64, q0, 32);

  float lm[8];
#pragma unroll
  for (int r = 0; r < 8; ++r) lm[r] = -3.0e30f;
  __syncthreads();

  // scores: S = Q.K^T + qdot-gather, scaled by 1/sqrt(D)=0.125
  for (int kt = 0; kt < 64; ++kt) {
    int k0 = kt * 16;
    v8f c = {};
    c = WMMA_F16(qa0, load_B_f16(kp, 64, k0, 0),  c);
    c = WMMA_F16(qa1, load_B_f16(kp, 64, k0, 32), c);
#pragma unroll
    for (int r = 0; r < 8; ++r) {
      int M = r + 8 * grp;
      int q = q0 + M, k = k0 + N;
      int dist = q - k;
      dist = dist < -128 ? -128 : (dist > 128 ? 128 : dist);
      float s = (c[r] + qdl[M][dist + 128]) * 0.125f;
      Pb[M][k0 + N] = (_Float16)s;
      lm[r] = fmaxf(lm[r], s);
    }
  }
  // row max across the 16 lanes of each half-wave
#pragma unroll
  for (int r = 0; r < 8; ++r) {
    float v = lm[r];
    for (int msk = 8; msk >= 1; msk >>= 1) v = fmaxf(v, __shfl_xor(v, msk, 32));
    lm[r] = v;
  }
  if (N == 0) {
#pragma unroll
    for (int r = 0; r < 8; ++r) rmax[r + 8 * grp] = lm[r];
  }
  __syncthreads();

  // exp + row sums (unnormalized exp kept in Pb)
  for (int r = 0; r < 16; ++r) {
    float mx = rmax[r], s = 0.f;
    for (int k = lane; k < 1024; k += 32) {
      float e = __expf((float)Pb[r][k] - mx);
      Pb[r][k] = (_Float16)e;
      s += e;
    }
    for (int msk = 16; msk >= 1; msk >>= 1) s += __shfl_xor(s, msk, 32);
    if (lane == 0) rsum[r] = s;
  }
  __syncthreads();

  // relative-distance buckets: pw[r][j] (normalized)
  for (int r = 0; r < 16; ++r) {
    int q = q0 + r;
    float inv = 1.0f / rsum[r];
    for (int j = 1 + lane; j < 256; j += 32) {
      int k = q - j + 128;
      float v = (k >= 0 && k < 1024) ? (float)Pb[r][k] : 0.f;
      pwb[r][j] = (_Float16)(v * inv);
    }
    float t0 = 0.f, t1 = 0.f;
    for (int k = q + 128 + lane; k < 1024; k += 32) t0 += (float)Pb[r][k];  // dist <= -128
    for (int k = lane; k <= q - 128; k += 32)       t1 += (float)Pb[r][k];  // dist >= 128
    for (int msk = 16; msk >= 1; msk >>= 1) {
      t0 += __shfl_xor(t0, msk, 32);
      t1 += __shfl_xor(t1, msk, 32);
    }
    if (lane == 0) { pwb[r][0] = (_Float16)(t0 * inv); pwb[r][256] = (_Float16)(t1 * inv); }
    for (int j = 257 + lane; j < 288; j += 32) pwb[r][j] = (_Float16)0.f;
  }
  __syncthreads();

  // w1 = softmax(P) . V  (contraction over k, f32 accumulate)
  float invm = 1.0f / rsum[N];  // A-frag row for this lane
  v8f z = {};
  v8f acc[4]; acc[0] = z; acc[1] = z; acc[2] = z; acc[3] = z;
  for (int kt = 0; kt < 32; ++kt) {
    int k0 = kt * 32;
    const _Float16* prow = &Pb[N][k0];
    v16h a;
#pragma unroll
    for (int e = 0; e < 8; ++e) a[e] = (_Float16)((float)prow[grp * 8 + e] * invm);
#pragma unroll
    for (int e = 0; e < 8; ++e) a[8 + e] = (_Float16)((float)prow[16 + grp * 8 + e] * invm);
#pragma unroll
    for (int dt = 0; dt < 4; ++dt)
      acc[dt] = WMMA_F16(a, load_B_f16_rm(vp, 64, k0, dt * 16), acc[dt]);
  }
  {
    float* o1 = w1buf + ((size_t)p * 1024 + q0) * 64;
#pragma unroll
    for (int dt = 0; dt < 4; ++dt)
#pragma unroll
      for (int r = 0; r < 8; ++r)
        o1[(size_t)(r + 8 * grp) * 64 + dt * 16 + N] = acc[dt][r];
  }

  // w2 = pw . rel_v  (contraction over padded 288 buckets)
  v8f acc2[4]; acc2[0] = z; acc2[1] = z; acc2[2] = z; acc2[3] = z;
  for (int jt = 0; jt < 9; ++jt) {
    int j0 = jt * 32;
    v16h a = load_A_f16(&pwb[0][0], 288, 0, j0);
#pragma unroll
    for (int dt = 0; dt < 4; ++dt)
      acc2[dt] = WMMA_F16(a, load_B_f16_rm(relv_h, 64, j0, dt * 16), acc2[dt]);
  }
  {
    float* o2 = w2buf + ((size_t)p2 * 1024 + q0) * 64;
#pragma unroll
    for (int dt = 0; dt < 4; ++dt)
#pragma unroll
      for (int r = 0; r < 8; ++r)
        o2[(size_t)(r + 8 * grp) * 64 + dt * 16 + N] = acc2[dt][r];
  }
}

// ---- K4: out = (w1+w2 gathered to (B,T,E)) @ Wo^T + bo ---------------------
// grid (128, 16) block 256.

__device__ inline v16h load_A_w(const float* __restrict__ w1, const float* __restrict__ w2,
                                int i0, int e0) {
  int lane = threadIdx.x & 31, grp = lane >> 4, mr = lane & 15;
  int i = i0 + mr, bb = i >> 10, t = i & 1023;
  v16h a;
  int eb = e0 + grp * 8;
  {
    int h = eb >> 6, d = eb & 63;
    size_t off = (((size_t)(bb * 16 + h) * 1024) + t) * 64 + d;
#pragma unroll
    for (int e = 0; e < 8; ++e) a[e] = (_Float16)(w1[off + e] + w2[off + e]);
  }
  eb = e0 + 16 + grp * 8;
  {
    int h = eb >> 6, d = eb & 63;
    size_t off = (((size_t)(bb * 16 + h) * 1024) + t) * 64 + d;
#pragma unroll
    for (int e = 0; e < 8; ++e) a[8 + e] = (_Float16)(w1[off + e] + w2[off + e]);
  }
  return a;
}

__global__ __launch_bounds__(256) void out_kernel(const float* __restrict__ w1buf,
                                                  const float* __restrict__ w2buf,
                                                  const _Float16* __restrict__ woh,
                                                  const float* __restrict__ bo,
                                                  float* __restrict__ out) {
  int wave = threadIdx.x >> 5;
  int i0 = blockIdx.x * 32 + (wave >> 2) * 16;
  int j0 = blockIdx.y * 64 + (wave & 3) * 16;
  v8f c = {};
  for (int e0 = 0; e0 < 1024; e0 += 32) {
    v16h a = load_A_w(w1buf, w2buf, i0, e0);
    v16h b = load_B_f16(woh, 1024, j0, e0);
    c = WMMA_F16(a, b, c);
  }
  int lane = threadIdx.x & 31, grp = lane >> 4, N = lane & 15;
  int j = j0 + N;
#pragma unroll
  for (int r = 0; r < 8; ++r) {
    int i = i0 + r + 8 * grp;
    out[(size_t)i * 1024 + j] = c[r] + bo[j];
  }
}

// ---------------------------------------------------------------------------

extern "C" void kernel_launch(void* const* d_in, const int* in_sizes, int n_in,
                              void* d_out, int out_size, void* d_ws, size_t ws_size,
                              hipStream_t stream) {
  const float* hidden = (const float*)d_in[0];
  const float* Wq = (const float*)d_in[1];  const float* bq = (const float*)d_in[2];
  const float* Wk = (const float*)d_in[3];  const float* bk = (const float*)d_in[4];
  const float* Wv = (const float*)d_in[5];  const float* bv = (const float*)d_in[6];
  const float* Wo = (const float*)d_in[7];  const float* bo = (const float*)d_in[8];
  const float* relk = (const float*)d_in[9];
  const float* relv = (const float*)d_in[10];

  char* ws = (char*)d_ws;
  size_t off = 0;
  auto alloc = [&](size_t bytes) -> void* {
    void* p = ws + off;
    off += (bytes + 255) & ~(size_t)255;
    return p;
  };
  _Float16* hh     = (_Float16*)alloc(4096ull * 1024 * 2);     // hidden f16
  _Float16* wqh    = (_Float16*)alloc(1024ull * 1024 * 2);
  _Float16* wkh    = (_Float16*)alloc(1024ull * 1024 * 2);
  _Float16* wvh    = (_Float16*)alloc(1024ull * 1024 * 2);
  _Float16* woh    = (_Float16*)alloc(1024ull * 1024 * 2);
  _Float16* relk_h = (_Float16*)alloc(288ull * 64 * 2);        // padded table
  _Float16* relv_h = (_Float16*)alloc(288ull * 64 * 2);
  _Float16* qbuf   = (_Float16*)alloc(64ull * 1024 * 64 * 2);  // (pair,t,d)
  _Float16* kbuf   = (_Float16*)alloc(64ull * 1024 * 64 * 2);
  _Float16* vbuf   = (_Float16*)alloc(64ull * 1024 * 64 * 2);
  float*    qdot   = (float*)   alloc(64ull * 1024 * 288 * 4);
  float*    w1buf  = (float*)   alloc(64ull * 1024 * 64 * 4);
  float*    w2buf  = (float*)   alloc(64ull * 1024 * 64 * 4);
  (void)ws_size; (void)in_sizes; (void)n_in; (void)out_size;

  // K0: conversions
  cvt_f32_f16<<<(4194304 + 255) / 256, 256, 0, stream>>>(hidden, hh, 4194304);
  cvt_f32_f16<<<(1048576 + 255) / 256, 256, 0, stream>>>(Wq, wqh, 1048576);
  cvt_f32_f16<<<(1048576 + 255) / 256, 256, 0, stream>>>(Wk, wkh, 1048576);
  cvt_f32_f16<<<(1048576 + 255) / 256, 256, 0, stream>>>(Wv, wvh, 1048576);
  cvt_f32_f16<<<(1048576 + 255) / 256, 256, 0, stream>>>(Wo, woh, 1048576);
  cvt_pad_table<<<(288 * 64 + 255) / 256, 256, 0, stream>>>(relk, relk_h, 257, 288, 64);
  cvt_pad_table<<<(288 * 64 + 255) / 256, 256, 0, stream>>>(relv, relv_h, 257, 288, 64);

  // K1: Q/K/V projections
  proj_kernel<<<dim3(128, 16, 3), 256, 0, stream>>>(hh, wqh, wkh, wvh, bq, bk, bv,
                                                    qbuf, kbuf, vbuf);
  // K2: qdot GEMM
  qdot_kernel<<<dim3(64, 18, 64), 32, 0, stream>>>(qbuf, relk_h, qdot);

  // K3: attention (scores + softmax + w1 + w2)
  attn_kernel<<<dim3(64, 64), 32, 0, stream>>>(qbuf, kbuf, vbuf, qdot, relv_h,
                                               w1buf, w2buf);

  // K4: output projection
  out_kernel<<<dim3(128, 16), 256, 0, stream>>>(w1buf, w2buf, woh, bo, (float*)d_out);
}
